// csLoss_50225347560089
// MI455X (gfx1250) — matrix operand order repaired and checked by reading, not compiled, not run on hardware
//
#include <hip/hip_runtime.h>
#include <math.h>

typedef __attribute__((ext_vector_type(16))) _Float16 v16h;
typedef __attribute__((ext_vector_type(8)))  _Float16 v8h;
typedef __attribute__((ext_vector_type(8)))  float    v8f;

#define NN    8192
#define DD    64
#define TILE  128
#define TPM   64          // tiles per matrix dim (8192/128)
#define TILES_PER_PAIR (TPM * TPM)   // 4096
#define LDSS  72          // LDS row stride in halves (64 + 8 pad; 144B = 9*16B)

// ---------------------------------------------------------------- prep: row norms
__global__ __launch_bounds__(256) void csloss_prep(const float* __restrict__ X,
                                                   const float* __restrict__ Z,
                                                   float* __restrict__ norms) {
    int id = blockIdx.x * 256 + threadIdx.x;       // 0 .. 16383
    const float* src = (id < NN) ? X : Z;
    int row = id & (NN - 1);
    const float4* p = (const float4*)(src + (size_t)row * DD);
    float s = 0.f;
#pragma unroll
    for (int i = 0; i < DD / 4; ++i) {
        float4 v = p[i];
        s += v.x * v.x + v.y * v.y + v.z * v.z + v.w * v.w;
    }
    norms[id] = s;
}

// ---------------------------------------------------------------- main fused kernel
__global__ __launch_bounds__(256) void csloss_main(const float* __restrict__ X,
                                                   const float* __restrict__ Z,
                                                   const float* __restrict__ norms,
                                                   float* __restrict__ partial) {
    __shared__ _Float16 As[TILE * LDSS];
    __shared__ _Float16 Bs[TILE * LDSS];
    __shared__ float a2s[TILE];
    __shared__ float b2s[TILE];
    __shared__ float red[256];

    const int tid  = threadIdx.x;
    const int bid  = blockIdx.x;
    const int pair = bid >> 12;                 // /4096 : 0=XX 1=ZZ 2=XZ
    const int t    = bid & (TILES_PER_PAIR - 1);
    const int tm   = t >> 6;
    const int tn   = t & (TPM - 1);

    // symmetry: skip lower triangle for XX/ZZ; double strict upper triangle
    if (pair < 2 && tn < tm) {
        if (tid == 0) partial[bid] = 0.f;
        return;
    }
    const float weight = (pair < 2 && tn > tm) ? 2.f : 1.f;

    const float* Ag = (pair == 1) ? Z : X;
    const float* Bg = (pair == 0) ? X : Z;
    const float* na = (pair == 1) ? (norms + NN) : norms;
    const float* nb = (pair == 0) ? norms : (norms + NN);

    const int rowBase = tm * TILE;
    const int colBase = tn * TILE;

    // ---- stage tiles into LDS as f16 (each thread: half a row = 32 floats)
    {
        int r = tid >> 1, h = tid & 1;
        const float4* srcA = (const float4*)(Ag + (size_t)(rowBase + r) * DD + h * 32);
        const float4* srcB = (const float4*)(Bg + (size_t)(colBase + r) * DD + h * 32);
        _Float16* dA = &As[r * LDSS + h * 32];
        _Float16* dB = &Bs[r * LDSS + h * 32];
#pragma unroll
        for (int i = 0; i < 8; ++i) {
            float4 va = srcA[i], vb = srcB[i];
            dA[i * 4 + 0] = (_Float16)va.x; dA[i * 4 + 1] = (_Float16)va.y;
            dA[i * 4 + 2] = (_Float16)va.z; dA[i * 4 + 3] = (_Float16)va.w;
            dB[i * 4 + 0] = (_Float16)vb.x; dB[i * 4 + 1] = (_Float16)vb.y;
            dB[i * 4 + 2] = (_Float16)vb.z; dB[i * 4 + 3] = (_Float16)vb.w;
        }
        if (tid < 128) a2s[tid] = na[rowBase + tid];
        else           b2s[tid - 128] = nb[colBase + (tid - 128)];
    }
    __syncthreads();

    // ---- wave tiling: 8 waves, each owns 32 rows x 64 cols (2x4 subtiles of 16x16)
    const int wave = tid >> 5;
    const int lane = tid & 31;
    const int wm   = wave & 3;      // row group: wm*32
    const int wn   = wave >> 2;     // col group: wn*64
    const int lrow = lane & 15;
    const int hi   = lane >> 4;

    v8f acc[2][4];
#pragma unroll
    for (int m = 0; m < 2; ++m)
#pragma unroll
        for (int n = 0; n < 4; ++n)
            acc[m][n] = (v8f){0.f, 0.f, 0.f, 0.f, 0.f, 0.f, 0.f, 0.f};

#pragma unroll
    for (int k0 = 0; k0 < DD; k0 += 32) {
        v16h a[2], b[4];
        // A fragment (ISA 16-bit A 16x32 layout): lane holds K = k0+8*hi+{0..7} and k0+16+8*hi+{0..7}
#pragma unroll
        for (int m = 0; m < 2; ++m) {
            int row = wm * 32 + m * 16 + lrow;
            const v8h lo = *(const v8h*)&As[row * LDSS + k0 + hi * 8];
            const v8h hh = *(const v8h*)&As[row * LDSS + k0 + 16 + hi * 8];
#pragma unroll
            for (int i = 0; i < 8; ++i) { a[m][i] = lo[i]; a[m][i + 8] = hh[i]; }
        }
        // B fragment (K x N): lane = column N, holds K = k0+8*hi+{0..7} and k0+16+8*hi+{0..7}
#pragma unroll
        for (int n = 0; n < 4; ++n) {
            int col = wn * 64 + n * 16 + lrow;
            const v8h lo = *(const v8h*)&Bs[col * LDSS + k0 + hi * 8];
            const v8h hh = *(const v8h*)&Bs[col * LDSS + k0 + 16 + hi * 8];
#pragma unroll
            for (int i = 0; i < 8; ++i) { b[n][i] = lo[i]; b[n][i + 8] = hh[i]; }
        }
#pragma unroll
        for (int m = 0; m < 2; ++m)
#pragma unroll
            for (int n = 0; n < 4; ++n)
                acc[m][n] = __builtin_amdgcn_wmma_f32_16x16x32_f16(
                    false, a[m], false, b[n], (short)0, acc[m][n], false, false);
    }

    // ---- fused RBF epilogue + per-thread accumulation
    float lsum = 0.f;
#pragma unroll
    for (int m = 0; m < 2; ++m) {
        int rbase = wm * 32 + m * 16 + hi * 8;
#pragma unroll
        for (int n = 0; n < 4; ++n) {
            int col = wn * 64 + n * 16 + lrow;
            float b2v = b2s[col];
#pragma unroll
            for (int r = 0; r < 8; ++r) {
                float c  = acc[m][n][r];
                float d2 = a2s[rbase + r] + b2v - 2.f * c;
                d2 = fmaxf(d2, 0.f);
                lsum += __expf(-d2);           // KSIZE == 1.0
            }
        }
    }
    lsum *= weight;

    // ---- deterministic block reduction -> per-block partial
    red[tid] = lsum;
    __syncthreads();
#pragma unroll
    for (int s = 128; s > 0; s >>= 1) {
        if (tid < s) red[tid] += red[tid + s];
        __syncthreads();
    }
    if (tid == 0) partial[bid] = red[0];
}

// ---------------------------------------------------------------- finalize
__global__ __launch_bounds__(256) void csloss_finalize(const float* __restrict__ partial,
                                                       float* __restrict__ out) {
    __shared__ float sm[256];
    __shared__ float totals[3];
    const int tid = threadIdx.x;
    for (int p = 0; p < 3; ++p) {
        float acc = 0.f;
        for (int i = tid; i < TILES_PER_PAIR; i += 256)
            acc += partial[p * TILES_PER_PAIR + i];
        sm[tid] = acc;
        __syncthreads();
        for (int s = 128; s > 0; s >>= 1) {
            if (tid < s) sm[tid] += sm[tid + s];
            __syncthreads();
        }
        if (tid == 0) totals[p] = sm[0];
        __syncthreads();
    }
    if (tid == 0) {
        const float SCALE = 0.3989422804014327f / ((float)NN * (float)NN); // 1/sqrt(2*pi)/N^2
        float mxx = totals[0] * SCALE;
        float mzz = totals[1] * SCALE;
        float mxz = totals[2] * SCALE;
        out[0] = logf(sqrtf(mxx * mzz + 1e-5f) / (mxz + 1e-5f));
    }
}

// ---------------------------------------------------------------- launcher
extern "C" void kernel_launch(void* const* d_in, const int* in_sizes, int n_in,
                              void* d_out, int out_size, void* d_ws, size_t ws_size,
                              hipStream_t stream) {
    (void)in_sizes; (void)n_in; (void)out_size; (void)ws_size;
    const float* X = (const float*)d_in[0];
    const float* Z = (const float*)d_in[1];
    float* ws      = (float*)d_ws;
    float* norms   = ws;                 // 2*8192 floats
    float* partial = ws + 2 * NN;        // 3*4096 floats

    csloss_prep<<<(2 * NN) / 256, 256, 0, stream>>>(X, Z, norms);
    csloss_main<<<3 * TILES_PER_PAIR, 256, 0, stream>>>(X, Z, norms, partial);
    csloss_finalize<<<1, 256, 0, stream>>>(partial, (float*)d_out);
}
